// HPCNWithTemporalDynamics_28071906246885
// MI455X (gfx1250) — matrix operand order repaired and checked by reading, not compiled, not run on hardware
//
#include <hip/hip_runtime.h>
#include <hip/hip_bf16.h>

typedef __attribute__((ext_vector_type(16))) __bf16 v16bf;
typedef __attribute__((ext_vector_type(8)))  __bf16 v8bf;
typedef __attribute__((ext_vector_type(8)))  float  v8f;

#define B_  64
#define S_  512
#define H_  1024
#define L_  3
#define W_  5

// softmax([5,4,3,2,1]) precomputed
__device__ __constant__ float c_wsoft[W_] = {
    0.63640866f, 0.23412165f, 0.08612854f, 0.03168492f, 0.01165623f};

// ---------------- zero the temporal memory ring ----------------
__global__ __launch_bounds__(256) void zero_f32(float* p, int n) {
    int i = blockIdx.x * 256 + threadIdx.x;
    if (i < n) p[i] = 0.0f;
}

// ---------------- convert weights f32 -> bf16 (once) ----------------
__global__ __launch_bounds__(256) void cvt_w(const float* __restrict__ w,
                                             __bf16* __restrict__ wb, int n) {
    int i = blockIdx.x * 256 + threadIdx.x;
    if (i < n) wb[i] = (__bf16)w[i];
}

// ---------------- prep: cur = input + 0.1*ctx, store bf16 ----------------
// act_bf16 [64,1024]; mem ring [L][B][W][H]; rep_f32 [64,1024]
__global__ __launch_bounds__(256) void prep(const float* __restrict__ x,
                                            const float* __restrict__ rep,
                                            const float* __restrict__ mem,
                                            __bf16* __restrict__ act,
                                            int t, int layer) {
    int idx = blockIdx.x * 256 + threadIdx.x;      // 65536 threads
    int b = idx >> 10;
    int f = idx & (H_ - 1);
    float base = (layer == 0) ? x[(b * S_ + t) * H_ + f] : rep[idx];
    const float* m = mem + ((size_t)(layer * B_ + b) * W_) * H_ + f;
    float ctx = 0.0f;
#pragma unroll
    for (int w = 0; w < W_; ++w) {
        int slot = (t - 1 - w + 10) % W_;          // ring pos written at time t-1-w
        ctx += c_wsoft[w] * m[slot * H_];
    }
    act[idx] = (__bf16)(base + 0.1f * ctx);
}

// ---------------- WMMA GEMM: rep = tanh(act @ W^T + b) ----------------
// One wave per 16x16 output tile. 256 tiles = 32 blocks x 8 waves.
// A = act [64,1024] bf16 row-major. B(k,n) = W[n,k] -> read W rows.
__global__ __launch_bounds__(256) void gemm_tanh(const __bf16* __restrict__ A,
                                                 const __bf16* __restrict__ Wl,
                                                 const float* __restrict__ bias,
                                                 float* __restrict__ rep,
                                                 float* __restrict__ mem,
                                                 float* __restrict__ outp,
                                                 int t, int layer) {
    const int lane = threadIdx.x & 31;
    const int wave = threadIdx.x >> 5;
    const int tile = blockIdx.x * 8 + wave;        // 0..255
    const int m0 = (tile & 3) * 16;                // 4 M tiles
    const int n0 = (tile >> 2) * 16;               // 64 N tiles

    const int half = lane >> 4;                    // 0: lanes 0-15, 1: lanes 16-31
    const int l15  = lane & 15;

    // A fragment address pieces (contiguous K, matching B):
    //   lanes 0-15 : row=l15, K = k0+0..15
    //   lanes 16-31: row=l15, K = k0+16..31
    const __bf16* arow = A + (size_t)(m0 + l15) * H_ + half * 16;
    // B fragment (32x16, column per lane):
    //   lanes 0-15 : col=n0+l15, K = k0+0..15 ; lanes 16-31: K = k0+16..31
    const __bf16* bcol = Wl + (size_t)(n0 + l15) * H_ + half * 16;

    v8f acc = {};
    union { v16bf v; v8bf h[2]; } a;
#pragma unroll 4
    for (int k0 = 0; k0 < H_; k0 += 32) {
        const v8bf* ap = (const v8bf*)(arow + k0);
        a.h[0] = ap[0];                            // K = base+0..7
        a.h[1] = ap[1];                            // K = base+8..15
        v16bf bfr = *(const v16bf*)(bcol + k0);
        acc = __builtin_amdgcn_wmma_f32_16x16x32_bf16(
            false, a.v, false, bfr, (short)0, acc, false, false);
    }

    // Epilogue. C/D layout: lanes 0-15 -> M=m0+r, lanes 16-31 -> M=m0+8+r; N=n0+l15.
    const int nn = n0 + l15;
    const float bv = bias[nn];
    const int slot = t % W_;
#pragma unroll
    for (int r = 0; r < 8; ++r) {
        const int mm = m0 + half * 8 + r;
        float v = tanhf(acc[r] + bv);
        rep[(size_t)mm * H_ + nn] = v;
        mem[((size_t)(layer * B_ + mm) * W_ + slot) * H_ + nn] = v;
        if (outp) outp[((size_t)mm * S_ + t) * H_ + nn] = v;
    }
}

extern "C" void kernel_launch(void* const* d_in, const int* in_sizes, int n_in,
                              void* d_out, int out_size, void* d_ws, size_t ws_size,
                              hipStream_t stream) {
    const float* x  = (const float*)d_in[0];   // [64,512,1024]
    const float* Ws = (const float*)d_in[1];   // [3,1024,1024]
    const float* bs = (const float*)d_in[2];   // [3,1024]
    float* out = (float*)d_out;                // [64,512,1024]

    char* ws = (char*)d_ws;
    __bf16* Wbf   = (__bf16*)(ws);                               // 6 MB
    __bf16* act   = (__bf16*)(ws + (size_t)L_ * H_ * H_ * 2);    // 128 KB
    float*  rep   = (float*) (ws + (size_t)L_ * H_ * H_ * 2 + (size_t)B_ * H_ * 2);
    float*  mem   = (float*) ((char*)rep + (size_t)B_ * H_ * 4); // [3][64][5][1024]

    const int memN = L_ * B_ * W_ * H_;
    zero_f32<<<(memN + 255) / 256, 256, 0, stream>>>(mem, memN);
    const int wN = L_ * H_ * H_;
    cvt_w<<<(wN + 255) / 256, 256, 0, stream>>>(Ws, Wbf, wN);

    for (int t = 0; t < S_; ++t) {
        for (int l = 0; l < L_; ++l) {
            prep<<<(B_ * H_) / 256, 256, 0, stream>>>(x, rep, mem, act, t, l);
            gemm_tanh<<<32, 256, 0, stream>>>(
                act, Wbf + (size_t)l * H_ * H_, bs + l * H_,
                rep, mem, (l == L_ - 1) ? out : nullptr, t, l);
        }
    }
}
